// NeRFRenderer_87153476371107
// MI455X (gfx1250) — matrix operand order repaired and checked by reading, not compiled, not run on hardware
//
#include <hip/hip_runtime.h>
#include <hip/hip_bf16.h>
#include <math.h>

typedef __attribute__((ext_vector_type(16))) _Float16 v16h;
typedef __attribute__((ext_vector_type(8)))  _Float16 v8h;
typedef __attribute__((ext_vector_type(8)))  float    v8f;

#define N_RAYS    4096
#define N_SAMPLES 256
#define C_APP     27
#define FEAT      128

// -------- combine two 16B half-vectors into one WMMA fragment --------
template <typename P>
__device__ inline v16h ld_frag(const P* p0, const P* p1) {
  v8h a = *(const v8h*)p0;
  v8h b = *(const v8h*)p1;
  return __builtin_shufflevector(a, b, 0,1,2,3,4,5,6,7,8,9,10,11,12,13,14,15);
}

// ===================== kernel 1: weight conversion =====================
// w1T: [128 n][32 k] f16 (K padded 27->32), w2T: [128 n][128 k],
// w3T: [16 n][128 k] (N padded 3->16)
__global__ void prep_weights(const float* __restrict__ W1,
                             const float* __restrict__ W2,
                             const float* __restrict__ W3,
                             _Float16* __restrict__ w1T,
                             _Float16* __restrict__ w2T,
                             _Float16* __restrict__ w3T) {
  int idx = blockIdx.x * 256 + threadIdx.x;
  if (idx < 4096) {
    int n = idx >> 5, k = idx & 31;
    float v = (k < C_APP) ? W1[k * FEAT + n] : 0.f;      // W1 is [66][128]
    w1T[idx] = (_Float16)v;
  } else if (idx < 4096 + 16384) {
    int i = idx - 4096;
    int n = i >> 7, k = i & 127;
    w2T[i] = (_Float16)W2[k * FEAT + n];                  // W2 is [128][128]
  } else if (idx < 4096 + 16384 + 2048) {
    int i = idx - (4096 + 16384);
    int n = i >> 7, k = i & 127;
    w3T[i] = (_Float16)((n < 3) ? W3[k * 3 + n] : 0.f);   // W3 is [128][3]
  }
}

// ===================== kernel 2: per-ray constant term =====================
// rayterm[r][n] = b1[n] + sum over the 39 view-dependent inputs * W1 rows 27..65
__global__ void prep_rayterm(const float* __restrict__ viewdirs,
                             const float* __restrict__ W1,
                             const float* __restrict__ b1,
                             float* __restrict__ rayterm) {
  int r = blockIdx.x;
  int n = threadIdx.x;
  float vd[3];
  vd[0] = viewdirs[r * 3 + 0];
  vd[1] = viewdirs[r * 3 + 1];
  vd[2] = viewdirs[r * 3 + 2];
  float acc = b1[n];
  #pragma unroll
  for (int c = 0; c < 3; c++) acc += vd[c] * W1[(27 + c) * FEAT + n];
  #pragma unroll
  for (int c = 0; c < 3; c++) {
    float f = 1.f;
    #pragma unroll
    for (int k = 0; k < 6; k++) {
      float p = vd[c] * f;
      f *= 2.f;
      float s, co;
      __sincosf(p, &s, &co);
      acc += s  * W1[(30 + c * 6 + k) * FEAT + n];  // sin block at 27+3
      acc += co * W1[(48 + c * 6 + k) * FEAT + n];  // cos block at 27+3+18
    }
  }
  rayterm[r * FEAT + n] = acc;
}

// ===================== kernel 3: WMMA MLP =====================
// Block = 256 threads (8 waves), 128 contiguous points (all same ray).
// Wave w owns rows [w*16, w*16+16).
__launch_bounds__(256)
__global__ void mlp_kernel(const float* __restrict__ app,
                           const _Float16* __restrict__ w1T,
                           const _Float16* __restrict__ w2T,
                           const _Float16* __restrict__ w3T,
                           const float* __restrict__ rayterm,
                           const float* __restrict__ b2,
                           const float* __restrict__ b3,
                           float* __restrict__ rgbBuf) {
  __shared__ __align__(16) _Float16 sApp[128 * 32];   // [row][k] padded K
  __shared__ __align__(16) _Float16 sH[128 * 128];    // hidden activations

  const int tid = threadIdx.x;
  const int p0  = blockIdx.x * 128;
  const int ray = p0 >> 8;

  // ---- stage app features (f32 -> f16, K padded to 32) ----
  for (int i = tid; i < 128 * 32; i += 256) {
    int row = i >> 5, k = i & 31;
    float v = (k < C_APP) ? app[(size_t)(p0 + row) * C_APP + k] : 0.f;
    sApp[i] = (_Float16)v;
  }
  __syncthreads();

  const int lane = tid & 31;
  const int w    = tid >> 5;
  const int col  = lane & 15;
  const int hi   = lane >> 4;      // 0: lanes 0-15, 1: lanes 16-31

  // ---- layer 1: [16x32] x [32x128], one K-step per N tile ----
  const int aRow = w * 16 + col;
  v16h af = ld_frag(&sApp[aRow * 32 + hi * 8], &sApp[aRow * 32 + hi * 8 + 16]);

  v8f acc[8];
  #pragma unroll
  for (int n = 0; n < 8; n++) {
    float rt = rayterm[ray * FEAT + n * 16 + col];      // broadcast C (incl. b1)
    v8f c;
    #pragma unroll
    for (int v = 0; v < 8; v++) c[v] = rt;
    const _Float16* bp = &w1T[(n * 16 + col) * 32 + hi * 8];
    v16h bf = ld_frag(bp, bp + 16);
    acc[n] = __builtin_amdgcn_wmma_f32_16x16x32_f16(false, af, false, bf,
                                                    (short)0, c, false, false);
  }
  // ReLU + stash h1 into LDS in [row][128] f16
  const int rowBase = w * 16 + hi * 8;
  #pragma unroll
  for (int n = 0; n < 8; n++)
    #pragma unroll
    for (int v = 0; v < 8; v++) {
      float x = acc[n][v];
      sH[(rowBase + v) * FEAT + n * 16 + col] = (_Float16)(x > 0.f ? x : 0.f);
    }
  __syncthreads();

  // ---- layer 2: [16x128] x [128x128], 4 K-steps x 8 N tiles ----
  v8f acc2[8];
  #pragma unroll
  for (int n = 0; n < 8; n++) {
    float bb = b2[n * 16 + col];
    #pragma unroll
    for (int v = 0; v < 8; v++) acc2[n][v] = bb;
  }
  #pragma unroll
  for (int kc = 0; kc < 4; kc++) {
    const _Float16* ap = &sH[(w * 16 + col) * FEAT + kc * 32 + hi * 8];
    v16h a2 = ld_frag(ap, ap + 16);
    #pragma unroll
    for (int n = 0; n < 8; n++) {
      const _Float16* bp = &w2T[(n * 16 + col) * FEAT + kc * 32 + hi * 8];
      v16h bf = ld_frag(bp, bp + 16);
      acc2[n] = __builtin_amdgcn_wmma_f32_16x16x32_f16(false, a2, false, bf,
                                                       (short)0, acc2[n], false, false);
    }
  }
  __syncthreads();
  #pragma unroll
  for (int n = 0; n < 8; n++)
    #pragma unroll
    for (int v = 0; v < 8; v++) {
      float x = acc2[n][v];
      sH[(rowBase + v) * FEAT + n * 16 + col] = (_Float16)(x > 0.f ? x : 0.f);
    }
  __syncthreads();

  // ---- layer 3: [16x128] x [128x16] (cols 0..2 valid) ----
  v8f acc3;
  {
    float bb = (col < 3) ? b3[col] : 0.f;
    #pragma unroll
    for (int v = 0; v < 8; v++) acc3[v] = bb;
  }
  #pragma unroll
  for (int kc = 0; kc < 4; kc++) {
    const _Float16* ap = &sH[(w * 16 + col) * FEAT + kc * 32 + hi * 8];
    v16h a3 = ld_frag(ap, ap + 16);
    const _Float16* bp = &w3T[col * FEAT + kc * 32 + hi * 8];
    v16h bf = ld_frag(bp, bp + 16);
    acc3 = __builtin_amdgcn_wmma_f32_16x16x32_f16(false, a3, false, bf,
                                                  (short)0, acc3, false, false);
  }
  // sigmoid + write rgb
  if (col < 3) {
    #pragma unroll
    for (int v = 0; v < 8; v++) {
      int p = p0 + rowBase + v;
      float x = acc3[v];
      rgbBuf[(size_t)p * 3 + col] = 1.f / (1.f + __expf(-x));
    }
  }
}

// ===================== kernel 4: alpha composite (1 wave / ray) =====================
__launch_bounds__(256)
__global__ void composite(const float* __restrict__ density,
                          const float* __restrict__ dists,
                          const float* __restrict__ rgbBuf,
                          float* __restrict__ out) {
  const int lane = threadIdx.x & 31;
  const int w    = threadIdx.x >> 5;
  const int r    = blockIdx.x * 8 + w;
  const int sbase = r * N_SAMPLES + lane * 8;

  float a[8], t[8];
  float prod = 1.f;
  #pragma unroll
  for (int j = 0; j < 8; j++) {
    float sg = density[sbase + j];
    sg = sg > 0.f ? sg : 0.f;                       // relu
    float al = 1.f - __expf(-sg * dists[sbase + j] * 25.f);
    float tj = 1.f - al + 1e-10f;
    a[j] = al; t[j] = tj;
    prod *= tj;
  }
  // inclusive scan of per-lane transmittance products (wave32)
  float inc = prod;
  #pragma unroll
  for (int off = 1; off < 32; off <<= 1) {
    float v = __shfl_up(inc, off, 32);
    if (lane >= off) inc *= v;
  }
  float ex = __shfl_up(inc, 1, 32);
  if (lane == 0) ex = 1.f;

  float T = ex, wsum = 0.f, r0 = 0.f, r1 = 0.f, r2 = 0.f;
  #pragma unroll
  for (int j = 0; j < 8; j++) {
    float wj = a[j] * T;
    T *= t[j];
    wsum += wj;
    size_t b = (size_t)(sbase + j) * 3;
    r0 += wj * rgbBuf[b + 0];
    r1 += wj * rgbBuf[b + 1];
    r2 += wj * rgbBuf[b + 2];
  }
  #pragma unroll
  for (int off = 16; off > 0; off >>= 1) {
    wsum += __shfl_xor(wsum, off, 32);
    r0   += __shfl_xor(r0, off, 32);
    r1   += __shfl_xor(r1, off, 32);
    r2   += __shfl_xor(r2, off, 32);
  }
  if (lane == 0) {
    float bg = 1.f - wsum;
    out[r * 3 + 0] = fminf(fmaxf(r0 + bg, 0.f), 1.f);
    out[r * 3 + 1] = fminf(fmaxf(r1 + bg, 0.f), 1.f);
    out[r * 3 + 2] = fminf(fmaxf(r2 + bg, 0.f), 1.f);
  }
}

extern "C" void kernel_launch(void* const* d_in, const int* in_sizes, int n_in,
                              void* d_out, int out_size, void* d_ws, size_t ws_size,
                              hipStream_t stream) {
  const float* density  = (const float*)d_in[0];
  const float* app      = (const float*)d_in[1];
  const float* viewdirs = (const float*)d_in[2];
  const float* dists    = (const float*)d_in[3];
  const float* W1       = (const float*)d_in[4];
  const float* b1       = (const float*)d_in[5];
  const float* W2       = (const float*)d_in[6];
  const float* b2       = (const float*)d_in[7];
  const float* W3       = (const float*)d_in[8];
  const float* b3       = (const float*)d_in[9];

  char* ws = (char*)d_ws;
  _Float16* w1T   = (_Float16*)ws;                 // 4096 halves
  _Float16* w2T   = w1T + 4096;                    // 16384 halves
  _Float16* w3T   = w2T + 16384;                   // 2048 halves
  float* rayterm  = (float*)(ws + 64 * 1024);                    // 2 MB
  float* rgbBuf   = (float*)(ws + 64 * 1024 + 2 * 1024 * 1024);  // 12 MB

  prep_weights<<<88, 256, 0, stream>>>(W1, W2, W3, w1T, w2T, w3T);
  prep_rayterm<<<N_RAYS, FEAT, 0, stream>>>(viewdirs, W1, b1, rayterm);
  mlp_kernel<<<(N_RAYS * N_SAMPLES) / 128, 256, 0, stream>>>(
      app, w1T, w2T, w3T, rayterm, b2, b3, rgbBuf);
  composite<<<N_RAYS / 8, 256, 0, stream>>>(density, dists, rgbBuf, (float*)d_out);
}
